// GraphNet_90847148245714
// MI455X (gfx1250) — matrix operand order repaired and checked by reading, not compiled, not run on hardware
//
#include <hip/hip_runtime.h>
#include <hip/hip_bf16.h>

#define N_NODES 100000
#define N_EDGES 1280000
#define HDIM    64
#define H2      128
#define FNODE   16
#define FPOS    3
#define NLAYER  4
#define OUTC    3
#define OUT2    6

#define GEN_EPS 1e-7f
#define BN_EPS  1e-5f
#define LN_EPS  1e-5f
#define SLOPE   0.01f

typedef _Float16 half_t;
typedef __attribute__((ext_vector_type(16))) _Float16 v16h;
typedef __attribute__((ext_vector_type(8)))  float    v8f;

// ---------------- utility ----------------
__global__ void k_fill(float* __restrict__ p, int count, float v) {
    int i = blockIdx.x * blockDim.x + threadIdx.x;
    if (i < count) p[i] = v;
}

// ---------------- initial embedding: h = x@xw + xb + pos@pw + pb ----------------
__global__ void k_embed(const float* __restrict__ x, const float* __restrict__ pos,
                        const float* __restrict__ xw, const float* __restrict__ xb,
                        const float* __restrict__ pw, const float* __restrict__ pb,
                        float* __restrict__ h) {
    int tid = blockIdx.x * blockDim.x + threadIdx.x;
    if (tid >= N_NODES * HDIM) return;
    int n = tid >> 6, c = tid & 63;
    float acc = xb[c] + pb[c];
#pragma unroll
    for (int f = 0; f < FNODE; ++f) acc += x[n * FNODE + f] * xw[f * HDIM + c];
#pragma unroll
    for (int p = 0; p < FPOS; ++p)  acc += pos[n * FPOS + p] * pw[p * HDIM + c];
    h[tid] = acc;
}

// ---------------- global graph-LayerNorm stats (sum, sumsq) ----------------
__global__ void k_ln_reduce(const float* __restrict__ h, float* __restrict__ stats, int count) {
    __shared__ float s0[256], s1[256];
    float a = 0.f, b = 0.f;
    for (int i = blockIdx.x * blockDim.x + threadIdx.x; i < count; i += gridDim.x * blockDim.x) {
        float v = h[i]; a += v; b += v * v;
    }
    s0[threadIdx.x] = a; s1[threadIdx.x] = b; __syncthreads();
    for (int s = 128; s > 0; s >>= 1) {
        if ((int)threadIdx.x < s) { s0[threadIdx.x] += s0[threadIdx.x + s]; s1[threadIdx.x] += s1[threadIdx.x + s]; }
        __syncthreads();
    }
    if (threadIdx.x == 0) { atomicAdd(&stats[0], s0[0]); atomicAdd(&stats[1], s1[0]); }
}

__global__ void k_ln_fin(float* __restrict__ stats) {
    const float M = (float)N_NODES * HDIM;
    float mean = stats[0] / M;
    float var  = stats[1] / M - mean * mean;
    float sd   = sqrtf(fmaxf(var, 0.f));
    stats[2] = mean;
    stats[3] = 1.f / (sd + LN_EPS);
}

// ---------------- t = LeakyReLU(LN(h)*g + b) ----------------
__global__ void k_make_t(const float* __restrict__ h, const float* __restrict__ stats,
                         const float* __restrict__ g, const float* __restrict__ b,
                         float* __restrict__ t) {
    int tid = blockIdx.x * blockDim.x + threadIdx.x;
    if (tid >= N_NODES * HDIM) return;
    int c = tid & 63;
    float v = (h[tid] - stats[2]) * stats[3] * g[c] + b[c];
    t[tid] = (v >= 0.f) ? v : SLOPE * v;
}

// ---------------- edge pass 1: per-dst per-channel max of msg = relu(t[src])+eps ----
// msg > 0 always => integer atomicMax on float bit patterns is order-preserving (init 0).
__global__ void k_edge_max(const int* __restrict__ ei, const float* __restrict__ t,
                           float* __restrict__ mx) {
    int tid = blockIdx.x * blockDim.x + threadIdx.x;
    if (tid >= N_EDGES * 16) return;
    int e = tid >> 4, cg = (tid & 15) * 4;
    int src = ei[e], dst = ei[N_EDGES + e];
    const float4 tv = *(const float4*)(t + src * HDIM + cg);
    int* mp = (int*)(mx + dst * HDIM + cg);
    atomicMax(mp + 0, __float_as_int(fmaxf(tv.x, 0.f) + GEN_EPS));
    atomicMax(mp + 1, __float_as_int(fmaxf(tv.y, 0.f) + GEN_EPS));
    atomicMax(mp + 2, __float_as_int(fmaxf(tv.z, 0.f) + GEN_EPS));
    atomicMax(mp + 3, __float_as_int(fmaxf(tv.w, 0.f) + GEN_EPS));
}

// ---------------- edge pass 2: num += ex*msg, den += ex -------------------------
__global__ void k_edge_sum(const int* __restrict__ ei, const float* __restrict__ t,
                           const float* __restrict__ mx, float* __restrict__ num,
                           float* __restrict__ den) {
    int tid = blockIdx.x * blockDim.x + threadIdx.x;
    if (tid >= N_EDGES * 16) return;
    int e = tid >> 4, cg = (tid & 15) * 4;
    int src = ei[e], dst = ei[N_EDGES + e];
    const float4 tv = *(const float4*)(t + src * HDIM + cg);
    const float4 mv = *(const float4*)(mx + dst * HDIM + cg);
    float m0 = fmaxf(tv.x, 0.f) + GEN_EPS, m1 = fmaxf(tv.y, 0.f) + GEN_EPS;
    float m2 = fmaxf(tv.z, 0.f) + GEN_EPS, m3 = fmaxf(tv.w, 0.f) + GEN_EPS;
    float e0 = __expf(m0 - mv.x), e1 = __expf(m1 - mv.y);
    float e2 = __expf(m2 - mv.z), e3 = __expf(m3 - mv.w);
    float* dp = den + dst * HDIM + cg;
    float* np = num + dst * HDIM + cg;
    atomicAdd(dp + 0, e0); atomicAdd(dp + 1, e1); atomicAdd(dp + 2, e2); atomicAdd(dp + 3, e3);
    atomicAdd(np + 0, e0 * m0); atomicAdd(np + 1, e1 * m1);
    atomicAdd(np + 2, e2 * m2); atomicAdd(np + 3, e3 * m3);
}

// ---------------- GEMM1 (WMMA): h1 = (num/den + t) @ c1w + c1b; BN partial sums --
__global__ __launch_bounds__(256)
void k_gemm1(const float* __restrict__ num, const float* __restrict__ den,
             const float* __restrict__ t, const float* __restrict__ w,   // [64,128]
             const float* __restrict__ bias,                             // [128]
             float* __restrict__ h1, float* __restrict__ bnsum, float* __restrict__ bnsq) {
    const int wid  = threadIdx.x >> 5;
    const int lane = threadIdx.x & 31;
    const int tile = blockIdx.x * 8 + wid;
    if (tile >= (N_NODES / 16)) return;   // wave-uniform: EXEC stays all-ones
    const int m0   = tile * 16;
    const int mrow = m0 + (lane & 15);
    const int hi   = lane >> 4;
    const int col  = lane & 15;

    v16h a[2];
#pragma unroll
    for (int kt = 0; kt < 2; ++kt) {
        const int k0 = kt * 32 + hi * 8;
#pragma unroll
        for (int e = 0; e < 16; ++e) {
            int k = k0 + e + ((e >= 8) ? 8 : 0);
            int idx = mrow * HDIM + k;
            float d = den[idx];
            float v = ((d > 0.f) ? num[idx] / d : 0.f) + t[idx];
            a[kt][e] = (half_t)v;
        }
    }
#pragma unroll
    for (int nt = 0; nt < 8; ++nt) {
        const int n0 = nt * 16;
        v8f acc = {};
#pragma unroll
        for (int kt = 0; kt < 2; ++kt) {
            const int kb = kt * 32 + hi * 8;
            v16h b;
#pragma unroll
            for (int e = 0; e < 16; ++e)
                b[e] = (half_t)w[(kb + e + ((e >= 8) ? 8 : 0)) * H2 + (n0 + col)];
            acc = __builtin_amdgcn_wmma_f32_16x16x32_f16(
                false, a[kt], false, b, (short)0, acc, false, false);
        }
        float bs = bias[n0 + col];
        float s = 0.f, s2 = 0.f;
#pragma unroll
        for (int r = 0; r < 8; ++r) {
            float v = acc[r] + bs;
            h1[(m0 + r + hi * 8) * H2 + (n0 + col)] = v;
            s += v; s2 += v * v;
        }
        atomicAdd(&bnsum[n0 + col], s);
        atomicAdd(&bnsq[n0 + col], s2);
    }
}

// ---------------- BN finalize: scale/shift per channel ----------------
__global__ void k_bn_fin(const float* __restrict__ sum, const float* __restrict__ sq,
                         const float* __restrict__ g, const float* __restrict__ b,
                         float* __restrict__ scale, float* __restrict__ shift, int C) {
    int c = blockIdx.x * blockDim.x + threadIdx.x;
    if (c >= C) return;
    const float invN = 1.f / (float)N_NODES;
    float mu  = sum[c] * invN;
    float var = sq[c] * invN - mu * mu;
    float sc  = g[c] * rsqrtf(fmaxf(var, 0.f) + BN_EPS);
    scale[c] = sc;
    shift[c] = b[c] - mu * sc;
}

// ---------------- GEMM2 (WMMA): h += relu(BN(h1)) @ c2w + c2b ----------------
__global__ __launch_bounds__(256)
void k_gemm2(const float* __restrict__ h1, const float* __restrict__ scale,
             const float* __restrict__ shift, const float* __restrict__ w,  // [128,64]
             const float* __restrict__ bias,                                // [64]
             float* __restrict__ h) {
    const int wid  = threadIdx.x >> 5;
    const int lane = threadIdx.x & 31;
    const int tile = blockIdx.x * 8 + wid;
    if (tile >= (N_NODES / 16)) return;
    const int m0   = tile * 16;
    const int mrow = m0 + (lane & 15);
    const int hi   = lane >> 4;
    const int col  = lane & 15;

    v16h a[4];
#pragma unroll
    for (int kt = 0; kt < 4; ++kt) {
        const int k0 = kt * 32 + hi * 8;
#pragma unroll
        for (int e = 0; e < 16; ++e) {
            int k = k0 + e + ((e >= 8) ? 8 : 0);
            float v = fmaxf(h1[mrow * H2 + k] * scale[k] + shift[k], 0.f);
            a[kt][e] = (half_t)v;
        }
    }
#pragma unroll
    for (int nt = 0; nt < 4; ++nt) {
        const int n0 = nt * 16;
        v8f acc = {};
#pragma unroll
        for (int kt = 0; kt < 4; ++kt) {
            const int kb = kt * 32 + hi * 8;
            v16h b;
#pragma unroll
            for (int e = 0; e < 16; ++e)
                b[e] = (half_t)w[(kb + e + ((e >= 8) ? 8 : 0)) * HDIM + (n0 + col)];
            acc = __builtin_amdgcn_wmma_f32_16x16x32_f16(
                false, a[kt], false, b, (short)0, acc, false, false);
        }
        float bs = bias[n0 + col];
#pragma unroll
        for (int r = 0; r < 8; ++r)
            h[(m0 + r + hi * 8) * HDIM + (n0 + col)] += acc[r] + bs;
    }
}

// ---------------- output head ----------------
__global__ void k_ho(const float* __restrict__ h, const float* __restrict__ ow,
                     const float* __restrict__ ob, float* __restrict__ ho) {
    int n = blockIdx.x * blockDim.x + threadIdx.x;
    if (n >= N_NODES) return;
    float a0 = ob[0], a1 = ob[1], a2 = ob[2];
#pragma unroll 8
    for (int k = 0; k < HDIM; ++k) {
        float v = h[n * HDIM + k];
        a0 += v * ow[k * 3 + 0]; a1 += v * ow[k * 3 + 1]; a2 += v * ow[k * 3 + 2];
    }
    ho[n * 3 + 0] = a0; ho[n * 3 + 1] = a1; ho[n * 3 + 2] = a2;
}

__global__ void k_edge_max3(const int* __restrict__ ei, const float* __restrict__ ho,
                            float* __restrict__ mx3) {
    int e = blockIdx.x * blockDim.x + threadIdx.x;
    if (e >= N_EDGES) return;
    int src = ei[e], dst = ei[N_EDGES + e];
#pragma unroll
    for (int j = 0; j < 3; ++j) {
        float m = fmaxf(ho[src * 3 + j], 0.f) + GEN_EPS;
        atomicMax((int*)&mx3[dst * 3 + j], __float_as_int(m));
    }
}

__global__ void k_edge_sum3(const int* __restrict__ ei, const float* __restrict__ ho,
                            const float* __restrict__ mx3, float* __restrict__ num3,
                            float* __restrict__ den3) {
    int e = blockIdx.x * blockDim.x + threadIdx.x;
    if (e >= N_EDGES) return;
    int src = ei[e], dst = ei[N_EDGES + e];
#pragma unroll
    for (int j = 0; j < 3; ++j) {
        float m  = fmaxf(ho[src * 3 + j], 0.f) + GEN_EPS;
        float ex = __expf(m - mx3[dst * 3 + j]);
        atomicAdd(&den3[dst * 3 + j], ex);
        atomicAdd(&num3[dst * 3 + j], ex * m);
    }
}

__global__ void k_out_mlp1(const float* __restrict__ num3, const float* __restrict__ den3,
                           const float* __restrict__ ho, const float* __restrict__ w,  // [3,6]
                           const float* __restrict__ b, float* __restrict__ h1o,
                           float* __restrict__ bnsum, float* __restrict__ bnsq) {
    __shared__ float ls[OUT2], lq[OUT2];
    if (threadIdx.x < OUT2) { ls[threadIdx.x] = 0.f; lq[threadIdx.x] = 0.f; }
    __syncthreads();
    int n = blockIdx.x * blockDim.x + threadIdx.x;
    if (n < N_NODES) {
        float a[3];
#pragma unroll
        for (int j = 0; j < 3; ++j) {
            float d = den3[n * 3 + j];
            a[j] = ((d > 0.f) ? num3[n * 3 + j] / d : 0.f) + ho[n * 3 + j];
        }
#pragma unroll
        for (int c = 0; c < OUT2; ++c) {
            float v = b[c];
#pragma unroll
            for (int j = 0; j < 3; ++j) v += a[j] * w[j * OUT2 + c];
            h1o[n * OUT2 + c] = v;
            atomicAdd(&ls[c], v);          // ds_add_f32 (LDS atomics)
            atomicAdd(&lq[c], v * v);
        }
    }
    __syncthreads();
    if (threadIdx.x < OUT2) {
        atomicAdd(&bnsum[threadIdx.x], ls[threadIdx.x]);
        atomicAdd(&bnsq[threadIdx.x],  lq[threadIdx.x]);
    }
}

__global__ void k_out_final(const float* __restrict__ h1o, const float* __restrict__ scale,
                            const float* __restrict__ shift, const float* __restrict__ w,  // [6,3]
                            const float* __restrict__ b, float* __restrict__ out) {
    int n = blockIdx.x * blockDim.x + threadIdx.x;
    if (n >= N_NODES) return;
    float a0 = b[0], a1 = b[1], a2 = b[2];
#pragma unroll
    for (int k = 0; k < OUT2; ++k) {
        float v = fmaxf(h1o[n * OUT2 + k] * scale[k] + shift[k], 0.f);
        a0 += v * w[k * 3 + 0]; a1 += v * w[k * 3 + 1]; a2 += v * w[k * 3 + 2];
    }
    out[n * 3 + 0] = a0; out[n * 3 + 1] = a1; out[n * 3 + 2] = a2;
}

// ---------------- host launcher ----------------
extern "C" void kernel_launch(void* const* d_in, const int* in_sizes, int n_in,
                              void* d_out, int out_size, void* d_ws, size_t ws_size,
                              hipStream_t stream) {
    (void)in_sizes; (void)n_in; (void)out_size; (void)ws_size;
    const float* x     = (const float*)d_in[0];
    const float* pos   = (const float*)d_in[1];
    const int*   ei    = (const int*)  d_in[2];
    const float* xw    = (const float*)d_in[3];
    const float* xb    = (const float*)d_in[4];
    const float* pw    = (const float*)d_in[5];
    const float* pb    = (const float*)d_in[6];
    const float* ln_g  = (const float*)d_in[7];
    const float* ln_b  = (const float*)d_in[8];
    const float* c1w   = (const float*)d_in[9];
    const float* c1b   = (const float*)d_in[10];
    const float* bn_g  = (const float*)d_in[11];
    const float* bn_b  = (const float*)d_in[12];
    const float* c2w   = (const float*)d_in[13];
    const float* c2b   = (const float*)d_in[14];
    const float* ow    = (const float*)d_in[15];
    const float* ob    = (const float*)d_in[16];
    const float* o1w   = (const float*)d_in[17];
    const float* o1b   = (const float*)d_in[18];
    const float* obn_g = (const float*)d_in[19];
    const float* obn_b = (const float*)d_in[20];
    const float* o2w   = (const float*)d_in[21];
    const float* o2b   = (const float*)d_in[22];
    float* out = (float*)d_out;

    // workspace layout (floats)
    float* W = (float*)d_ws;
    float* h    = W;
    float* t    = h    + (size_t)N_NODES * HDIM;
    float* num  = t    + (size_t)N_NODES * HDIM;
    float* den  = num  + (size_t)N_NODES * HDIM;
    float* h1   = den  + (size_t)N_NODES * HDIM;   // N*128; mx aliases into it (disjoint live ranges)
    float* mx   = h1;
    float* ho   = h1   + (size_t)N_NODES * H2;
    float* num3 = ho   + (size_t)N_NODES * 3;
    float* den3 = num3 + (size_t)N_NODES * 3;
    float* mx3  = den3 + (size_t)N_NODES * 3;
    float* h1o  = mx3  + (size_t)N_NODES * 3;
    float* st   = h1o  + (size_t)N_NODES * OUT2;   // 1024-float stats block
    // st[0]=ln_sum st[1]=ln_sq st[2]=mean st[3]=inv ; st+4 bn_sum ; st+132 bn_sq ; st+260 scale ; st+388 shift

    const int NH  = N_NODES * HDIM;
    const int gNH = (NH + 255) / 256;
    const int gE16 = (N_EDGES * 16 + 255) / 256;
    const int gE   = (N_EDGES + 255) / 256;
    const int gTile = (N_NODES / 16 + 7) / 8;
    const int gN   = (N_NODES + 255) / 256;

    k_embed<<<gNH, 256, 0, stream>>>(x, pos, xw, xb, pw, pb, h);

    for (int i = 0; i < NLAYER; ++i) {
        k_fill<<<(516 + 255) / 256, 256, 0, stream>>>(st, 516, 0.f);
        k_fill<<<gNH, 256, 0, stream>>>(mx,  NH, 0.f);
        k_fill<<<gNH, 256, 0, stream>>>(num, NH, 0.f);
        k_fill<<<gNH, 256, 0, stream>>>(den, NH, 0.f);

        k_ln_reduce<<<1024, 256, 0, stream>>>(h, st, NH);
        k_ln_fin<<<1, 1, 0, stream>>>(st);
        k_make_t<<<gNH, 256, 0, stream>>>(h, st, ln_g + i * HDIM, ln_b + i * HDIM, t);

        k_edge_max<<<gE16, 256, 0, stream>>>(ei, t, mx);
        k_edge_sum<<<gE16, 256, 0, stream>>>(ei, t, mx, num, den);

        k_gemm1<<<gTile, 256, 0, stream>>>(num, den, t,
            c1w + (size_t)i * HDIM * H2, c1b + i * H2, h1, st + 4, st + 132);
        k_bn_fin<<<1, 128, 0, stream>>>(st + 4, st + 132,
            bn_g + i * H2, bn_b + i * H2, st + 260, st + 388, H2);
        k_gemm2<<<gTile, 256, 0, stream>>>(h1, st + 260, st + 388,
            c2w + (size_t)i * H2 * HDIM, c2b + i * HDIM, h);
    }

    // output head: ho = h@ow + ob, then 3-channel GENConv
    k_ho<<<gN, 256, 0, stream>>>(h, ow, ob, ho);
    k_fill<<<(516 + 255) / 256, 256, 0, stream>>>(st, 516, 0.f);
    k_fill<<<(N_NODES * 3 + 255) / 256, 256, 0, stream>>>(mx3,  N_NODES * 3, 0.f);
    k_fill<<<(N_NODES * 3 + 255) / 256, 256, 0, stream>>>(num3, N_NODES * 3, 0.f);
    k_fill<<<(N_NODES * 3 + 255) / 256, 256, 0, stream>>>(den3, N_NODES * 3, 0.f);

    k_edge_max3<<<gE, 256, 0, stream>>>(ei, ho, mx3);
    k_edge_sum3<<<gE, 256, 0, stream>>>(ei, ho, mx3, num3, den3);
    k_out_mlp1<<<gN, 256, 0, stream>>>(num3, den3, ho, o1w, o1b, h1o, st + 4, st + 132);
    k_bn_fin<<<1, 32, 0, stream>>>(st + 4, st + 132, obn_g, obn_b, st + 260, st + 388, OUT2);
    k_out_final<<<gN, 256, 0, stream>>>(h1o, st + 260, st + 388, o2w, o2b, out);
}
